// SA_56169582297373
// MI455X (gfx1250) — matrix operand order repaired and checked by reading, not compiled, not run on hardware
//
#include <hip/hip_runtime.h>

typedef __attribute__((ext_vector_type(16))) __bf16 v16bf;
typedef __attribute__((ext_vector_type(8)))  float  v8f;

constexpr int cB = 16, cC = 256, cN = 4096, cH = 64, cW = 64;
constexpr int cHEADS = 4, cKK = 16, cVV = 64;

// ---------------------------------------------------------------------------
// Kernel 0: pack weights (wq|wk|wv -> 9 o-tiles) and embedding into WMMA
// A-fragment order (bf16). A-frag 16x32 layout: lane l (m=l&15, g=l>>4),
// element e: K = g*16+e (contiguous, same order as the B-frags).
// ---------------------------------------------------------------------------
__global__ void k_pack(const float* __restrict__ wq, const float* __restrict__ wk,
                       const float* __restrict__ wv, const float* __restrict__ emb,
                       __bf16* __restrict__ wA, __bf16* __restrict__ embA) {
  int tid = blockIdx.x * blockDim.x + threadIdx.x;
  int stride = gridDim.x * blockDim.x;
  const int total = 9 * 8 * 32 * 16;  // otile, kstep, lane, elem
  for (int i = tid; i < total; i += stride) {
    int e = i & 15;
    int lane = (i >> 4) & 31;
    int kstep = (i >> 9) & 7;
    int otile = i >> 12;
    int m = lane & 15, g = lane >> 4;
    int kl = g * 16 + e;
    int c = kstep * 32 + kl;
    float val;
    if (otile < 4)       val = wq[(otile * 16 + m) * cC + c];
    else if (otile == 4) val = wk[m * cC + c];
    else                 val = wv[((otile - 5) * 16 + m) * cC + c];
    wA[i] = (__bf16)val;
  }
  const int etotal = 2 * 32 * 16;  // kstep(2: taps 0-31 / 32-63), lane, elem
  for (int i = tid; i < etotal; i += stride) {
    int e = i & 15;
    int lane = (i >> 4) & 31;
    int kstep = i >> 9;
    int m = lane & 15, g = lane >> 4;
    int kl = g * 16 + e;
    int t = kstep * 32 + kl;  // tap index (dy*7+dx), 49 valid taps
    float val = (t < 49) ? emb[m * 49 + t] : 0.0f;
    embA[i] = (__bf16)val;
  }
}

// ---------------------------------------------------------------------------
// Kernel 1: transpose x [B,C,N] f32 -> xT [B,N,C] bf16 (LDS 32x32 tile)
// ---------------------------------------------------------------------------
__global__ void k_transpose(const float* __restrict__ x, __bf16* __restrict__ xT) {
  __shared__ float tile[32][33];
  int b = blockIdx.z;
  int c0 = blockIdx.y * 32;
  int n0 = blockIdx.x * 32;
  int tid = threadIdx.x;           // 256
  int col = tid & 31, row8 = tid >> 5;
  const float* xb = x + (size_t)b * cC * cN;
#pragma unroll
  for (int q = 0; q < 4; q++) {
    int r = row8 + q * 8;
    tile[r][col] = xb[(size_t)(c0 + r) * cN + n0 + col];
  }
  __syncthreads();
  __bf16* xTb = xT + (size_t)b * cN * cC;
#pragma unroll
  for (int q = 0; q < 4; q++) {
    int r = row8 + q * 8;
    xTb[(size_t)(n0 + r) * cC + c0 + col] = (__bf16)tile[col][r];
  }
}

// ---------------------------------------------------------------------------
// Kernel 2: projection GEMM via WMMA bf16, B-fragment register reuse.
// grid (b, ntile 0..63), block 128 = 4 waves; each wave holds its 8 B-frags
// (K=C=256) in VGPRs and loops over all 9 o-tiles -> 72 WMMAs per wave.
// B-frag 32x16: lane l (n=l&15, g=l>>4), element e: K = g*16+e (contiguous c).
// ---------------------------------------------------------------------------
__global__ void k_proj(const __bf16* __restrict__ xT, const __bf16* __restrict__ wA,
                       float* __restrict__ Qraw, float* __restrict__ Kraw,
                       float* __restrict__ Vraw) {
  int b = blockIdx.x;
  int ntile = blockIdx.y;
  int wave = threadIdx.x >> 5;
  int lane = threadIdx.x & 31;
  int g = lane >> 4;
  int ncol = ntile * 64 + wave * 16 + (lane & 15);

  const __bf16* xrow = xT + ((size_t)b * cN + ncol) * cC + g * 16;
  v16bf bfrag[8];
#pragma unroll
  for (int ks = 0; ks < 8; ks++) bfrag[ks] = *(const v16bf*)(xrow + ks * 32);

  for (int otile = 0; otile < 9; otile++) {
    const __bf16* wbase = wA + (size_t)otile * 8 * 512 + lane * 16;
    v8f acc = {};
#pragma unroll
    for (int ks = 0; ks < 8; ks++) {
      v16bf a = *(const v16bf*)(wbase + ks * 512);
      acc = __builtin_amdgcn_wmma_f32_16x16x32_bf16(false, a, false, bfrag[ks],
                                                    (short)0, acc, false, false);
    }
    float* out;
    int obase, nch;
    if (otile < 4)       { out = Qraw; obase = otile * 16; nch = 64; }
    else if (otile == 4) { out = Kraw; obase = 0;          nch = 16; }
    else                 { out = Vraw; obase = (otile - 5) * 16; nch = 64; }
    size_t base = (size_t)b * nch * cN;
#pragma unroll
    for (int r = 0; r < 8; r++) {
      int row = obase + r + 8 * g;        // C/D layout: M = r + 8*(lane>=16)
      out[base + (size_t)row * cN + ncol] = acc[r];
    }
  }
}

// ---------------------------------------------------------------------------
// Kernel 3: GroupNorm stats. grid (b, 36): 16 q-groups, 4 k-groups, 16 v-groups
// (4 channels x 4096 each). Writes mean and rsqrt(var+eps).
// ---------------------------------------------------------------------------
__global__ void k_stats(const float* __restrict__ Qraw, const float* __restrict__ Kraw,
                        const float* __restrict__ Vraw,
                        float* muq, float* rsq, float* muk, float* rsk,
                        float* muv, float* rsv) {
  int b = blockIdx.x, gi = blockIdx.y;
  const float* src; int ch0; float* muo; float* rso;
  if (gi < 16)      { src = Qraw + (size_t)b * 64 * cN; ch0 = gi * 4;
                      muo = muq + b * 16 + gi;        rso = rsq + b * 16 + gi; }
  else if (gi < 20) { src = Kraw + (size_t)b * 16 * cN; ch0 = (gi - 16) * 4;
                      muo = muk + b * 4 + (gi - 16);  rso = rsk + b * 4 + (gi - 16); }
  else              { src = Vraw + (size_t)b * 64 * cN; ch0 = (gi - 20) * 4;
                      muo = muv + b * 16 + (gi - 20); rso = rsv + b * 16 + (gi - 20); }
  int tid = threadIdx.x;
  float s = 0.f, ss = 0.f;
  for (int i = tid; i < 4 * cN; i += 256) {
    float v = src[(size_t)(ch0 + (i >> 12)) * cN + (i & 4095)];
    s += v; ss += v * v;
  }
  __shared__ float sh[256], sh2[256];
  sh[tid] = s; sh2[tid] = ss; __syncthreads();
  for (int st = 128; st > 0; st >>= 1) {
    if (tid < st) { sh[tid] += sh[tid + st]; sh2[tid] += sh2[tid + st]; }
    __syncthreads();
  }
  if (tid == 0) {
    float mean = sh[0] * (1.0f / (4.0f * cN));
    float var = sh2[0] * (1.0f / (4.0f * cN)) - mean * mean;
    *muo = mean;
    *rso = rsqrtf(var + 1e-5f);
  }
}

// ---------------------------------------------------------------------------
// Kernel 4: k GroupNorm (on the fly) + softmax over N -> Ksm bf16.
// grid (b, ch 0..15), 256 threads x 16 elems in registers.
// ---------------------------------------------------------------------------
__global__ void k_softmax(const float* __restrict__ Kraw, const float* __restrict__ muk,
                          const float* __restrict__ rsk, const float* __restrict__ gkg,
                          const float* __restrict__ gkb, __bf16* __restrict__ Ksm) {
  int b = blockIdx.x, ch = blockIdx.y;
  int tid = threadIdx.x;
  int grp = ch >> 2;
  float scale = rsk[b * 4 + grp] * gkg[ch];
  float shift = gkb[ch] - muk[b * 4 + grp] * scale;
  const float* row = Kraw + ((size_t)b * 16 + ch) * cN;
  float vals[16];
  float mx = -3.0e38f;
#pragma unroll
  for (int i = 0; i < 16; i++) {
    vals[i] = row[tid + i * 256] * scale + shift;
    mx = fmaxf(mx, vals[i]);
  }
  __shared__ float sh[256];
  sh[tid] = mx; __syncthreads();
  for (int st = 128; st > 0; st >>= 1) {
    if (tid < st) sh[tid] = fmaxf(sh[tid], sh[tid + st]);
    __syncthreads();
  }
  mx = sh[0]; __syncthreads();
  float s = 0.f;
#pragma unroll
  for (int i = 0; i < 16; i++) { vals[i] = __expf(vals[i] - mx); s += vals[i]; }
  sh[tid] = s; __syncthreads();
  for (int st = 128; st > 0; st >>= 1) {
    if (tid < st) sh[tid] += sh[tid + st];
    __syncthreads();
  }
  float inv = 1.0f / sh[0];
  __bf16* orow = Ksm + ((size_t)b * 16 + ch) * cN;
#pragma unroll
  for (int i = 0; i < 16; i++) orow[tid + i * 256] = (__bf16)(vals[i] * inv);
}

// ---------------------------------------------------------------------------
// Kernel 5: lam_c[b][k][v] = sum_n Ksm[k,n] * Vn[v,n] via WMMA bf16.
// grid (b, vtile 0..3), 1 wave. A = Ksm (M=k, K=n), B = GN(V)^T (K=n, N=v).
// ---------------------------------------------------------------------------
__global__ void k_lamc(const __bf16* __restrict__ Ksm, const float* __restrict__ Vraw,
                       const float* __restrict__ muv, const float* __restrict__ rsv,
                       const float* __restrict__ gvg, const float* __restrict__ gvb,
                       float* __restrict__ lamc) {
  int b = blockIdx.x, vt = blockIdx.y;
  int lane = threadIdx.x & 31;
  int g = lane >> 4, m = lane & 15;
  int v = vt * 16 + m;
  float scale = rsv[b * 16 + (v >> 2)] * gvg[v];
  float shift = gvb[v] - muv[b * 16 + (v >> 2)] * scale;
  const __bf16* krow = Ksm + ((size_t)b * 16 + m) * cN;
  const float* vrow = Vraw + ((size_t)b * 64 + v) * cN;
  v8f acc = {};
  for (int ns = 0; ns < cN / 32; ns++) {
    v16bf a, bb;
#pragma unroll
    for (int e = 0; e < 16; e++) {
      int kl = g * 16 + e;
      a[e] = krow[ns * 32 + kl];
    }
#pragma unroll
    for (int e = 0; e < 16; e++) {
      int nl = g * 16 + e;
      bb[e] = (__bf16)(vrow[ns * 32 + nl] * scale + shift);
    }
    acc = __builtin_amdgcn_wmma_f32_16x16x32_bf16(false, a, false, bb,
                                                  (short)0, acc, false, false);
  }
#pragma unroll
  for (int r = 0; r < 8; r++) {
    int k = r + 8 * g;
    lamc[((size_t)b * 16 + k) * 64 + v] = acc[r];
  }
}

// ---------------------------------------------------------------------------
// Kernel 6: fused positional conv (WMMA) + final contraction, 8 v-channels
// per block. grid (y 0..63, vg 0..7, b 0..15), block 256 = 8 waves.
// Wave w owns v-channel v0+w: ctx[k,x] = sum_t emb[k,t]*vn[y+dy-3,x+dx-3]
// as 4 x-tiles x 2 K=32 WMMAs. GN'd q row-slab staged once in LDS; then
// out[h*64+v, n] = sum_k q[h,k,n]*(lamc[k,v] + ctx[k,v,n]).
// ---------------------------------------------------------------------------
__global__ void k_final(const float* __restrict__ Qraw, const float* __restrict__ Vraw,
                        const __bf16* __restrict__ embA, const float* __restrict__ lamc,
                        const float* __restrict__ muq, const float* __restrict__ rsq,
                        const float* __restrict__ muv, const float* __restrict__ rsv,
                        const float* __restrict__ gqg, const float* __restrict__ gqb,
                        const float* __restrict__ gvg, const float* __restrict__ gvb,
                        float* __restrict__ out) {
  int b = blockIdx.z;
  int v0 = blockIdx.y * 8;
  int y = blockIdx.x;
  int tid = threadIdx.x;           // 256
  int lane = tid & 31, wave = tid >> 5;

  __shared__ __bf16 vt_s[8][7][72];   // per v: 7 halo rows x 70 cols (padded)
  __shared__ float  q_s[64][65];      // [x][qc], GN applied
  __shared__ float  ctx_s[8][64][17]; // [v][x][k]
  __shared__ float  lamc_s[16][8];    // [k][v]
  __shared__ float  vsc_s[8], vsh_s[8];
  __shared__ float  qsc_s[64], qsh_s[64];

  // stage 0: per-channel affine params
  if (tid < 8) {
    int v = v0 + tid;
    float sc = rsv[b * 16 + (v >> 2)] * gvg[v];
    vsc_s[tid] = sc;
    vsh_s[tid] = gvb[v] - muv[b * 16 + (v >> 2)] * sc;
  } else if (tid >= 64 && tid < 128) {
    int qc = tid - 64;
    float sc = rsq[b * 16 + (qc >> 2)] * gqg[qc];
    qsc_s[qc] = sc;
    qsh_s[qc] = gqb[qc] - muq[b * 16 + (qc >> 2)] * sc;
  } else if (tid >= 128 && tid < 256) {
    int i = tid - 128;
    int k = i >> 3, j = i & 7;
    lamc_s[k][j] = lamc[((size_t)b * 16 + k) * 64 + v0 + j];
  }
  __syncthreads();

  // stage 1: v halo tiles (zero-padded borders, GN applied) -> LDS bf16
  for (int i = tid; i < 8 * 7 * 70; i += 256) {
    int vl = i / 490;
    int rem = i - vl * 490;
    int r = rem / 70, cc = rem - r * 70;
    int yy = y - 3 + r;
    int xx = cc - 3;
    float val = 0.0f;
    if (yy >= 0 && yy < cH && xx >= 0 && xx < cW)
      val = Vraw[((size_t)b * 64 + v0 + vl) * cN + yy * cW + xx] * vsc_s[vl] + vsh_s[vl];
    vt_s[vl][r][cc] = (__bf16)val;
  }
  // stage 2: q slab (64 x-positions, 64 channels) -> LDS f32
  for (int i = tid; i < 64 * 64; i += 256) {
    int x = i & 63, qc = i >> 6;
    q_s[x][qc] = Qraw[((size_t)b * 64 + qc) * cN + y * 64 + x] * qsc_s[qc] + qsh_s[qc];
  }
  __syncthreads();

  // stage 3: ctx via WMMA, wave w -> v-channel w, 4 x-tiles of 16
  {
    int g = lane >> 4;
    v16bf a0 = *(const v16bf*)(embA + lane * 16);
    v16bf a1 = *(const v16bf*)(embA + 512 + lane * 16);
#pragma unroll
    for (int xt = 0; xt < 4; xt++) {
      int x = xt * 16 + (lane & 15);
      v16bf b0, b1;
#pragma unroll
      for (int e = 0; e < 16; e++) {
        int t = g * 16 + e;                 // taps 0..31
        b0[e] = vt_s[wave][t / 7][x + t % 7];
      }
#pragma unroll
      for (int e = 0; e < 16; e++) {
        int t = 32 + g * 16 + e;            // taps 32..63 (>=49 padded)
        b1[e] = (t < 49) ? vt_s[wave][t / 7][x + t % 7] : (__bf16)0.0f;
      }
      v8f acc = {};
      acc = __builtin_amdgcn_wmma_f32_16x16x32_bf16(false, a0, false, b0,
                                                    (short)0, acc, false, false);
      acc = __builtin_amdgcn_wmma_f32_16x16x32_bf16(false, a1, false, b1,
                                                    (short)0, acc, false, false);
#pragma unroll
      for (int r = 0; r < 8; r++) ctx_s[wave][x][r + 8 * g] = acc[r];
    }
  }
  __syncthreads();

  // stage 4: out[h*64+v][n] = sum_k q[h,k,n]*(lamc+ctx); 8 outputs/thread
#pragma unroll
  for (int p = 0; p < 8; p++) {
    int idx = p * 256 + tid;
    int x = idx & 63;
    int rest = idx >> 6;       // 0..31
    int h = rest & 3;
    int vl = rest >> 2;        // 0..7
    float s = 0.0f;
#pragma unroll
    for (int k = 0; k < 16; k++) {
      s += q_s[x][h * 16 + k] * (lamc_s[k][vl] + ctx_s[vl][x][k]);
    }
    out[((size_t)b * 256 + h * 64 + v0 + vl) * cN + y * 64 + x] = s;
  }
}

// ---------------------------------------------------------------------------
extern "C" void kernel_launch(void* const* d_in, const int* in_sizes, int n_in,
                              void* d_out, int out_size, void* d_ws, size_t ws_size,
                              hipStream_t stream) {
  const float* x   = (const float*)d_in[0];
  const float* wq  = (const float*)d_in[1];
  const float* gqg = (const float*)d_in[2];
  const float* gqb = (const float*)d_in[3];
  const float* wk  = (const float*)d_in[4];
  const float* gkg = (const float*)d_in[5];
  const float* gkb = (const float*)d_in[6];
  const float* wv  = (const float*)d_in[7];
  const float* gvg = (const float*)d_in[8];
  const float* gvb = (const float*)d_in[9];
  const float* emb = (const float*)d_in[10];
  float* out = (float*)d_out;

  char* ws = (char*)d_ws;
  size_t off = 0;
  auto alloc = [&](size_t bytes) {
    void* p = ws + off;
    off = (off + bytes + 511) & ~(size_t)511;
    return p;
  };
  __bf16* wA   = (__bf16*)alloc((size_t)9 * 8 * 512 * 2);
  __bf16* embA = (__bf16*)alloc((size_t)2 * 512 * 2);
  __bf16* xT   = (__bf16*)alloc((size_t)cB * cN * cC * 2);
  float*  Qraw = (float*)alloc((size_t)cB * 64 * cN * 4);
  float*  Kraw = (float*)alloc((size_t)cB * 16 * cN * 4);
  float*  Vraw = (float*)alloc((size_t)cB * 64 * cN * 4);
  __bf16* Ksm  = (__bf16*)alloc((size_t)cB * 16 * cN * 2);
  float*  lamc = (float*)alloc((size_t)cB * 16 * 64 * 4);
  float*  muq  = (float*)alloc(cB * 16 * 4);
  float*  rsq  = (float*)alloc(cB * 16 * 4);
  float*  muk  = (float*)alloc(cB * 4 * 4);
  float*  rsk  = (float*)alloc(cB * 4 * 4);
  float*  muv  = (float*)alloc(cB * 16 * 4);
  float*  rsv  = (float*)alloc(cB * 16 * 4);

  k_pack<<<dim3(32), 256, 0, stream>>>(wq, wk, wv, emb, wA, embA);
  k_transpose<<<dim3(cN / 32, cC / 32, cB), 256, 0, stream>>>(x, xT);
  k_proj<<<dim3(cB, cN / 64), 128, 0, stream>>>(xT, wA, Qraw, Kraw, Vraw);
  k_stats<<<dim3(cB, 36), 256, 0, stream>>>(Qraw, Kraw, Vraw, muq, rsq, muk, rsk, muv, rsv);
  k_softmax<<<dim3(cB, 16), 256, 0, stream>>>(Kraw, muk, rsk, gkg, gkb, Ksm);
  k_lamc<<<dim3(cB, 4), 32, 0, stream>>>(Ksm, Vraw, muv, rsv, gvg, gvb, lamc);
  k_final<<<dim3(cH, cVV / 8, cB), 256, 0, stream>>>(Qraw, Vraw, embA, lamc,
                                                     muq, rsq, muv, rsv,
                                                     gqg, gqb, gvg, gvb, out);
}
